// TSSGC_10711648436454
// MI455X (gfx1250) — matrix operand order repaired and checked by reading, not compiled, not run on hardware
//
#include <hip/hip_runtime.h>
#include <math.h>

#define N_NODES 50000
#define N_EDGES 400000
#define DIM     128
#define HIDD    64
#define TDIM    32
#define HEADS   4

typedef __attribute__((ext_vector_type(16))) _Float16 v16h;
typedef __attribute__((ext_vector_type(8)))  float    v8f;

static __device__ __forceinline__ float lrelu02(float v) { return v > 0.f ? v : 0.2f * v; }
static __device__ __forceinline__ float sigmoidf_(float v) { return 1.f / (1.f + expf(-v)); }

// ------------------------------------------------------------------ fill
__global__ void k_fill_u32(unsigned* __restrict__ p, unsigned v, size_t n) {
  size_t i = (size_t)blockIdx.x * blockDim.x + threadIdx.x;
  if (i < n) p[i] = v;
}

// ------------------------------------------------------------------ time/amount stats (ddof=1)
__global__ void k_stats4(const float* __restrict__ t, const float* __restrict__ a,
                         int n, float* __restrict__ out) {
  __shared__ float s0[256], s1[256], s2[256], s3[256];
  float st = 0.f, st2 = 0.f, sa = 0.f, sa2 = 0.f;
  for (int i = threadIdx.x; i < n; i += blockDim.x) {
    float x = t[i]; st += x; st2 += x * x;
    float y = a[i]; sa += y; sa2 += y * y;
  }
  s0[threadIdx.x] = st; s1[threadIdx.x] = st2; s2[threadIdx.x] = sa; s3[threadIdx.x] = sa2;
  __syncthreads();
  for (int stp = 128; stp > 0; stp >>= 1) {
    if (threadIdx.x < stp) {
      s0[threadIdx.x] += s0[threadIdx.x + stp];
      s1[threadIdx.x] += s1[threadIdx.x + stp];
      s2[threadIdx.x] += s2[threadIdx.x + stp];
      s3[threadIdx.x] += s3[threadIdx.x + stp];
    }
    __syncthreads();
  }
  if (threadIdx.x == 0) {
    float mt = s0[0] / n;
    float vt = (s1[0] - n * mt * mt) / (n - 1);
    out[0] = mt; out[1] = 1.f / (sqrtf(fmaxf(vt, 0.f)) + 1e-6f);
    float ma = s2[0] / n;
    float va = (s3[0] - n * ma * ma) / (n - 1);
    out[2] = ma; out[3] = 1.f / (sqrtf(fmaxf(va, 0.f)) + 1e-6f);
  }
}

// ------------------------------------------------------------------ GRU(seq=1,h0=0) + temporal attention, one node per block
__global__ void k_temporal(const float* __restrict__ x, const float* __restrict__ timev,
                           const float* __restrict__ amount, const float* __restrict__ stat,
                           const float* __restrict__ Wih, const float* __restrict__ bih,
                           const float* __restrict__ bhh,
                           const float* __restrict__ taW1, const float* __restrict__ tab1,
                           const float* __restrict__ taW2, const float* __restrict__ tab2,
                           float* __restrict__ ht, int n) {
  int node = blockIdx.x;
  if (node >= n) return;
  __shared__ float xs[DIM + 2];
  __shared__ float gi[3 * TDIM];
  __shared__ float go[TDIM];
  __shared__ float t1[TDIM];
  __shared__ float awsh;
  int tid = threadIdx.x;
  if (tid < DIM) xs[tid] = x[(size_t)node * DIM + tid];
  if (tid == 0) {
    xs[DIM]     = (timev[node]  - stat[0]) * stat[1];
    xs[DIM + 1] = (amount[node] - stat[2]) * stat[3];
  }
  __syncthreads();
  if (tid < 3 * TDIM) {
    float acc = bih[tid];
    for (int k = 0; k < DIM + 2; ++k) acc += xs[k] * Wih[k * (3 * TDIM) + tid];
    gi[tid] = acc;
  }
  __syncthreads();
  if (tid < TDIM) {
    float r  = sigmoidf_(gi[tid] + bhh[tid]);
    float z  = sigmoidf_(gi[TDIM + tid] + bhh[TDIM + tid]);
    float nc = tanhf(gi[2 * TDIM + tid] + r * bhh[2 * TDIM + tid]);
    go[tid] = (1.f - z) * nc;
  }
  __syncthreads();
  if (tid < TDIM) {
    float acc = tab1[tid];
    for (int k = 0; k < TDIM; ++k) acc += go[k] * taW1[k * TDIM + tid];
    t1[tid] = tanhf(acc);
  }
  __syncthreads();
  if (tid == 0) {
    float s = tab2[0];
    for (int k = 0; k < TDIM; ++k) s += t1[k] * taW2[k];
    awsh = sigmoidf_(s);
  }
  __syncthreads();
  if (tid < TDIM) ht[(size_t)node * TDIM + tid] = go[tid] * awsh;
}

// ------------------------------------------------------------------ semantic MLP, one node per block (64 threads)
__global__ void k_semantic(const float* __restrict__ x, const int* __restrict__ ntype,
                           const float* __restrict__ emb, const float* __restrict__ W,
                           const float* __restrict__ b, float* __restrict__ hs, int n) {
  int node = blockIdx.x;
  if (node >= n) return;
  __shared__ float xs[DIM + HIDD];
  int tid = threadIdx.x;
  xs[tid]       = x[(size_t)node * DIM + tid];
  xs[64 + tid]  = x[(size_t)node * DIM + 64 + tid];
  xs[128 + tid] = emb[(size_t)ntype[node] * HIDD + tid];
  __syncthreads();
  float acc = b[tid];
  for (int k = 0; k < DIM + HIDD; ++k) acc += xs[k] * W[(size_t)k * HIDD + tid];
  hs[(size_t)node * HIDD + tid] = fmaxf(acc, 0.f);
}

// ------------------------------------------------------------------ GEMM via split-f16 WMMA (near-fp32 accuracy)
// C[M,Nc] = A[M,K] @ W[K,Nc] (+bias)(+relu). M%16==0, K%32==0, Nc%16==0.
// Block = 8 waves sharing one 16-col tile of W. Double-buffered LDS staging:
// the 32x16 f32 B tile for step s+1 is fetched with GLOBAL_LOAD_ASYNC_TO_LDS_B64
// (one 8B chunk per thread) while step s computes; synchronized with
// s_wait_asynccnt + one barrier per step. Row stride 18 dwords kills the
// lane-half bank conflict while keeping 8B alignment for the b64 async writes.
#define BSTRIDE 18
template<bool BIAS, bool RELU>
__global__ void __launch_bounds__(256)
k_gemm_wmma(const float* __restrict__ A, const float* __restrict__ W,
            const float* __restrict__ bias, float* __restrict__ C,
            int M, int K, int Nc) {
  __shared__ __align__(16) float smemB[2][32 * BSTRIDE];
  const int lane   = threadIdx.x & 31;
  const int wave   = threadIdx.x >> 5;
  const int tilesM = M >> 4;
  const int tn = blockIdx.x;                 // column tile (shared by all 8 waves)
  int tm = blockIdx.y * 8 + wave;            // row tile, private per wave
  const bool active = (tm < tilesM);
  if (!active) tm = 0;                       // clamp: must keep hitting barriers
  const int half = lane >> 4;                // 0: lanes 0-15, 1: lanes 16-31
  const int l16  = lane & 15;
  const size_t rowA = (size_t)(tm * 16 + l16);
  const int    colB = tn * 16 + l16;

  // cooperative B staging: thread -> 8B chunk (2 floats) of the 32x16 tile
  const int brow = threadIdx.x >> 3;         // 0..31
  const int bcol = (threadIdx.x & 7) * 2;    // 0,2,..,14
  // low 32 bits of the generic (flat) address of a __shared__ object == LDS offset
  const unsigned lds0 =
      (unsigned)(uintptr_t)(const void*)&smemB[0][brow * BSTRIDE + bcol];
  const unsigned lds1 =
      (unsigned)(uintptr_t)(const void*)&smemB[1][brow * BSTRIDE + bcol];

  const float* gbase = W + (size_t)brow * Nc + (tn * 16 + bcol);
  auto issue_copy = [&](int k0, unsigned ldso) {
    unsigned long long ga = (unsigned long long)(const void*)(gbase + (size_t)k0 * Nc);
    asm volatile("global_load_async_to_lds_b64 %0, %1, off"
                 :: "v"(ldso), "v"(ga) : "memory");
  };

  const int nsteps = K >> 5;
  issue_copy(0, lds0);
  asm volatile("s_wait_asynccnt 0x0" ::: "memory");
  __syncthreads();                           // buffer 0 ready for all waves

  v8f acc = {};
  for (int s = 0; s < nsteps; ++s) {
    if (s + 1 < nsteps) issue_copy((s + 1) << 5, (s & 1) ? lds0 : lds1);
    const int k0 = s << 5;
    // A frag (16x32 f16): lane half selects K base (0 or 8); elems 8..15 are K+16.
    v16h ah, al;
    const float* ap = A + rowA * K + k0 + half * 8;
#pragma unroll
    for (int i = 0; i < 8; ++i) {
      float v0 = ap[i];
      float v1 = ap[16 + i];
      _Float16 h0 = (_Float16)v0, h1 = (_Float16)v1;
      ah[i]     = h0; al[i]     = (_Float16)(v0 - (float)h0);
      ah[i + 8] = h1; al[i + 8] = (_Float16)(v1 - (float)h1);
    }
    // B frag (32x16 f16) from LDS: lane half selects K base (0 or 8); elems 8..15 are K+16.
    const float* bb = smemB[s & 1];
    v16h bh, bl;
    const int kb = half * 8;
#pragma unroll
    for (int j = 0; j < 16; ++j) {
      float v = bb[(kb + (j & 7) + ((j >> 3) << 4)) * BSTRIDE + l16];
      _Float16 h = (_Float16)v;
      bh[j] = h; bl[j] = (_Float16)(v - (float)h);
    }
    acc = __builtin_amdgcn_wmma_f32_16x16x32_f16(false, al, false, bh, (short)0, acc, false, false);
    acc = __builtin_amdgcn_wmma_f32_16x16x32_f16(false, ah, false, bl, (short)0, acc, false, false);
    acc = __builtin_amdgcn_wmma_f32_16x16x32_f16(false, ah, false, bh, (short)0, acc, false, false);
    // next tile's copy (issued above) has overlapped this step's compute
    asm volatile("s_wait_asynccnt 0x0" ::: "memory");
    __syncthreads();                         // all waves done reading buf[s&1]; buf[(s+1)&1] ready
  }
  if (active) {
    // C/D layout: VGPR i -> row (tm*16 + half*8 + i), col = tn*16 + (lane&15)
#pragma unroll
    for (int i = 0; i < 8; ++i) {
      int r = tm * 16 + half * 8 + i;
      float v = acc[i];
      if (BIAS) v += bias[colB];
      if (RELU) v = fmaxf(v, 0.f);
      C[(size_t)r * Nc + colB] = v;
    }
  }
}

// ------------------------------------------------------------------ attention logits: al_s/al_d [N,H]
__global__ void k_att_logits(const float* __restrict__ xw, const float* __restrict__ a_src,
                             const float* __restrict__ a_dst, float* __restrict__ alS,
                             float* __restrict__ alD, int n, int H, int C) {
  int idx = blockIdx.x * blockDim.x + threadIdx.x;
  if (idx >= n * H) return;
  int node = idx / H, h = idx - node * H;
  const float* row = xw + (size_t)node * H * C + (size_t)h * C;
  float ss = 0.f, sd = 0.f;
  for (int c = 0; c < C; ++c) {
    float v = row[c];
    ss += v * a_src[h * C + c];
    sd += v * a_dst[h * C + c];
  }
  alS[idx] = ss; alD[idx] = sd;
}

// monotone float<->uint mapping for order-invariant atomic max
static __device__ __forceinline__ unsigned f2u_mono(float f) {
  unsigned b = __float_as_uint(f);
  return (b & 0x80000000u) ? ~b : (b | 0x80000000u);
}
static __device__ __forceinline__ float u2f_mono(unsigned u) {
  return (u & 0x80000000u) ? __uint_as_float(u ^ 0x80000000u) : __uint_as_float(~u);
}

// ------------------------------------------------------------------ edge pass 1: segment max (incl self loops)
__global__ void k_edge_max(const int* __restrict__ src, const int* __restrict__ dst,
                           int E, int n, int H, const float* __restrict__ alS,
                           const float* __restrict__ alD, unsigned* __restrict__ emaxu) {
  int e = blockIdx.x * blockDim.x + threadIdx.x;
  if (e >= E + n) return;
  int s = (e < E) ? src[e] : (e - E);
  int d = (e < E) ? dst[e] : (e - E);
  for (int h = 0; h < H; ++h) {
    float v = lrelu02(alS[s * H + h] + alD[d * H + h]);
    atomicMax(&emaxu[d * H + h], f2u_mono(v));
  }
}

// ------------------------------------------------------------------ edge pass 2: exp + segment sum
__global__ void k_edge_exp(const int* __restrict__ src, const int* __restrict__ dst,
                           int E, int n, int H, const float* __restrict__ alS,
                           const float* __restrict__ alD, const unsigned* __restrict__ emaxu,
                           float* __restrict__ ex, float* __restrict__ denom) {
  int e = blockIdx.x * blockDim.x + threadIdx.x;
  if (e >= E + n) return;
  int s = (e < E) ? src[e] : (e - E);
  int d = (e < E) ? dst[e] : (e - E);
  for (int h = 0; h < H; ++h) {
    float v  = lrelu02(alS[s * H + h] + alD[d * H + h]);
    float m  = u2f_mono(emaxu[d * H + h]);
    float xv = expf(v - m);
    ex[(size_t)e * H + h] = xv;
    atomicAdd(&denom[d * H + h], xv);
  }
}

// ------------------------------------------------------------------ edge pass 3: weighted scatter (one block per edge)
__global__ void k_edge_agg(const int* __restrict__ src, const int* __restrict__ dst,
                           int E, int n, int H, int C, const float* __restrict__ xw,
                           const float* __restrict__ ex, const float* __restrict__ denom,
                           float* __restrict__ out) {
  int e = blockIdx.x;
  int s = (e < E) ? src[e] : (e - E);
  int d = (e < E) ? dst[e] : (e - E);
  int t = threadIdx.x;            // [0, H*C)
  int h = t / C;
  float w = ex[(size_t)e * H + h] / denom[d * H + h];
  atomicAdd(&out[(size_t)d * H * C + t], xw[(size_t)s * H * C + t] * w);
}

// ------------------------------------------------------------------ BatchNorm stats (biased var), one channel per block
__global__ void k_bn_stats(const float* __restrict__ h, int n, int nc,
                           float* __restrict__ mean, float* __restrict__ inv) {
  int ch = blockIdx.x;
  __shared__ float sh[256], sh2[256];
  float s = 0.f, s2 = 0.f;
  for (int r = threadIdx.x; r < n; r += blockDim.x) {
    float v = h[(size_t)r * nc + ch];
    s += v; s2 += v * v;
  }
  sh[threadIdx.x] = s; sh2[threadIdx.x] = s2;
  __syncthreads();
  for (int stp = 128; stp > 0; stp >>= 1) {
    if (threadIdx.x < stp) {
      sh[threadIdx.x]  += sh[threadIdx.x + stp];
      sh2[threadIdx.x] += sh2[threadIdx.x + stp];
    }
    __syncthreads();
  }
  if (threadIdx.x == 0) {
    float m = sh[0] / n;
    float v = sh2[0] / n - m * m;
    mean[ch] = m;
    inv[ch]  = rsqrtf(fmaxf(v, 0.f) + 1e-5f);
  }
}

// ------------------------------------------------------------------ BN apply (+optional ELU)
__global__ void k_bn_apply(const float* __restrict__ h, const float* __restrict__ mean,
                           const float* __restrict__ inv, const float* __restrict__ g,
                           const float* __restrict__ be, float* __restrict__ out,
                           int n, int nc, int do_elu) {
  size_t i = (size_t)blockIdx.x * blockDim.x + threadIdx.x;
  if (i >= (size_t)n * nc) return;
  int ch = (int)(i % nc);
  float y = (h[i] - mean[ch]) * inv[ch] * g[ch] + be[ch];
  if (do_elu) y = (y > 0.f) ? y : expm1f(y);
  out[i] = y;
}

// ------------------------------------------------------------------ concat [ht(32) | hsp(64) | hsem(64)] -> hc[N,160]
__global__ void k_concat(const float* __restrict__ ht, const float* __restrict__ hsp,
                         const float* __restrict__ hsem, float* __restrict__ hc, int n) {
  size_t i = (size_t)blockIdx.x * blockDim.x + threadIdx.x;
  if (i >= (size_t)n * 160) return;
  int node = (int)(i / 160), k = (int)(i - (size_t)node * 160);
  float v;
  if (k < 32)      v = ht[(size_t)node * 32 + k];
  else if (k < 96) v = hsp[(size_t)node * 64 + (k - 32)];
  else             v = hsem[(size_t)node * 64 + (k - 96)];
  hc[i] = v;
}

// ------------------------------------------------------------------ final 32->2 head + log_softmax
__global__ void k_head(const float* __restrict__ c1, const float* __restrict__ W2,
                       const float* __restrict__ b2, float* __restrict__ out, int n) {
  int node = blockIdx.x * blockDim.x + threadIdx.x;
  if (node >= n) return;
  float o0 = b2[0], o1 = b2[1];
  const float* row = c1 + (size_t)node * 32;
  for (int k = 0; k < 32; ++k) {
    float v = row[k];
    o0 += v * W2[k * 2];
    o1 += v * W2[k * 2 + 1];
  }
  float m = fmaxf(o0, o1);
  float lse = m + logf(expf(o0 - m) + expf(o1 - m));
  out[(size_t)node * 2]     = o0 - lse;
  out[(size_t)node * 2 + 1] = o1 - lse;
}

// ==================================================================
extern "C" void kernel_launch(void* const* d_in, const int* in_sizes, int n_in,
                              void* d_out, int out_size, void* d_ws, size_t ws_size,
                              hipStream_t stream) {
  (void)in_sizes; (void)n_in; (void)out_size; (void)ws_size;
  const float* x      = (const float*)d_in[0];
  const int*   src    = (const int*)d_in[1];
  const int*   dst    = (const int*)d_in[2];
  const float* timev  = (const float*)d_in[3];
  const float* amount = (const float*)d_in[4];
  const int*   ntype  = (const int*)d_in[5];
  const float* W0  = (const float*)d_in[6];
  const float* as0 = (const float*)d_in[7];
  const float* ad0 = (const float*)d_in[8];
  // b0/b1/b2 (d_in[9],[15],[21]) cancel exactly inside BatchNorm's mean-subtract -> skipped.
  const float* g0  = (const float*)d_in[10];
  const float* be0 = (const float*)d_in[11];
  const float* W1  = (const float*)d_in[12];
  const float* as1 = (const float*)d_in[13];
  const float* ad1 = (const float*)d_in[14];
  const float* g1  = (const float*)d_in[16];
  const float* be1 = (const float*)d_in[17];
  const float* W2  = (const float*)d_in[18];
  const float* as2 = (const float*)d_in[19];
  const float* ad2 = (const float*)d_in[20];
  const float* g2  = (const float*)d_in[22];
  const float* be2 = (const float*)d_in[23];
  const float* gW   = (const float*)d_in[24];
  const float* gbi  = (const float*)d_in[25];
  const float* gbh  = (const float*)d_in[26];
  const float* taW1 = (const float*)d_in[27];
  const float* tab1 = (const float*)d_in[28];
  const float* taW2 = (const float*)d_in[29];
  const float* tab2 = (const float*)d_in[30];
  const float* emb  = (const float*)d_in[31];
  const float* semW = (const float*)d_in[32];
  const float* semb = (const float*)d_in[33];
  const float* fW1  = (const float*)d_in[34];
  const float* fb1  = (const float*)d_in[35];
  const float* fW2  = (const float*)d_in[36];
  const float* fb2  = (const float*)d_in[37];
  const float* cW1  = (const float*)d_in[38];
  const float* cb1  = (const float*)d_in[39];
  const float* cW2  = (const float*)d_in[40];
  const float* cb2  = (const float*)d_in[41];
  float* out = (float*)d_out;

  const size_t n = N_NODES, e = N_EDGES, ep = e + n;
  const int tilesM = (int)(n / 16);             // 3125
  const unsigned gy = (unsigned)((tilesM + 7) / 8);
  float* ws = (float*)d_ws;
  size_t o = 0;
  float*    bufA  = ws + o; o += n * 256;        // GEMM outputs / xw / hc
  float*    bufB  = ws + o; o += n * 256;        // layer inputs / hf1
  float*    bufC  = ws + o; o += n * 256;        // aggregation accumulator / hf2+c1
  float*    alS   = ws + o; o += n * HEADS;
  float*    alD   = ws + o; o += n * HEADS;
  unsigned* emaxu = (unsigned*)(ws + o); o += n * HEADS;
  float*    denom = ws + o; o += n * HEADS;
  float*    exbuf = ws + o; o += ep * HEADS;
  float*    hT    = ws + o; o += n * TDIM;
  float*    hSem  = ws + o; o += n * HIDD;
  float*    hSp   = ws + o; o += n * HIDD;
  float*    stat4 = ws + o; o += 4;
  float*    bnM   = ws + o; o += 256;
  float*    bnI   = ws + o; o += 256;

  // ---- temporal & semantic branches (independent of GAT stack) ----
  k_stats4<<<1, 256, 0, stream>>>(timev, amount, (int)n, stat4);
  k_temporal<<<(unsigned)n, 128, 0, stream>>>(x, timev, amount, stat4, gW, gbi, gbh,
                                              taW1, tab1, taW2, tab2, hT, (int)n);
  k_semantic<<<(unsigned)n, 64, 0, stream>>>(x, ntype, emb, semW, semb, hSem, (int)n);

  // ---- one GAT layer ----
  auto run_gat = [&](const float* in, int K, const float* Wm, const float* as_,
                     const float* ad_, const float* g_, const float* be_,
                     int H, int C, float* normOut, int do_elu) {
    int Nc = H * C;
    k_gemm_wmma<false, false><<<dim3((unsigned)(Nc / 16), gy), 256, 0, stream>>>(
        in, Wm, nullptr, bufA, (int)n, K, Nc);
    int nh = (int)n * H;
    k_att_logits<<<(nh + 255) / 256, 256, 0, stream>>>(bufA, as_, ad_, alS, alD, (int)n, H, C);
    k_fill_u32<<<(nh + 255) / 256, 256, 0, stream>>>(emaxu, 0u, (size_t)nh);
    k_fill_u32<<<(nh + 255) / 256, 256, 0, stream>>>((unsigned*)denom, 0u, (size_t)nh);
    size_t nnc = n * (size_t)Nc;
    k_fill_u32<<<(unsigned)((nnc + 255) / 256), 256, 0, stream>>>((unsigned*)bufC, 0u, nnc);
    k_edge_max<<<(unsigned)((ep + 255) / 256), 256, 0, stream>>>(src, dst, (int)e, (int)n, H, alS, alD, emaxu);
    k_edge_exp<<<(unsigned)((ep + 255) / 256), 256, 0, stream>>>(src, dst, (int)e, (int)n, H, alS, alD, emaxu, exbuf, denom);
    k_edge_agg<<<(unsigned)ep, Nc, 0, stream>>>(src, dst, (int)e, (int)n, H, C, bufA, exbuf, denom, bufC);
    k_bn_stats<<<Nc, 256, 0, stream>>>(bufC, (int)n, Nc, bnM, bnI);
    k_bn_apply<<<(unsigned)((nnc + 255) / 256), 256, 0, stream>>>(bufC, bnM, bnI, g_, be_, normOut, (int)n, Nc, do_elu);
  };

  run_gat(x,    DIM,          W0, as0, ad0, g0, be0, HEADS, HIDD, bufB, 1);
  run_gat(bufB, HEADS * HIDD, W1, as1, ad1, g1, be1, HEADS, HIDD, bufB, 1);
  run_gat(bufB, HEADS * HIDD, W2, as2, ad2, g2, be2, 1,     HIDD, hSp,  0);

  // ---- fusion + classifier (bufA/bufB/bufC are free now; reuse) ----
  float* hc  = bufA;              // [N,160]
  float* hf1 = bufB;              // [N,128]
  float* hf2 = bufC;              // [N,64]
  float* c1  = bufC + n * 64;     // [N,32]
  size_t nhc = n * 160;
  k_concat<<<(unsigned)((nhc + 255) / 256), 256, 0, stream>>>(hT, hSp, hSem, hc, (int)n);
  k_gemm_wmma<true, true><<<dim3(128 / 16, gy), 256, 0, stream>>>(hc,  fW1, fb1, hf1, (int)n, 160, 128);
  k_gemm_wmma<true, true><<<dim3(64 / 16,  gy), 256, 0, stream>>>(hf1, fW2, fb2, hf2, (int)n, 128, 64);
  k_gemm_wmma<true, true><<<dim3(32 / 16,  gy), 256, 0, stream>>>(hf2, cW1, cb1, c1,  (int)n, 64,  32);
  k_head<<<(unsigned)((n + 255) / 256), 256, 0, stream>>>(c1, cW2, cb2, out, (int)n);
}
